// Encoder_8512625181341
// MI455X (gfx1250) — matrix operand order repaired and checked by reading, not compiled, not run on hardware
//
#include <hip/hip_runtime.h>
#include <hip/hip_bf16.h>
#include <math.h>

// ---------------- model constants (match reference) ----------------
#define BB   2
#define TT   2048
#define EE   1024
#define LL   4
#define HH   16
#define DHH  64
#define FEE  4096           // FX*E
#define VV   32000
#define BT   (BB*TT)        // 4096
#define E3   (3*EE)         // 3072

// Use CDNA5 async global->LDS DMA for GEMM tile staging (ASYNCcnt tracked).
#define GEMM_ASYNC 1

typedef __attribute__((ext_vector_type(16))) __bf16 v16bf;
typedef __attribute__((ext_vector_type(8)))  float  v8f;

struct BF16Frag { union { v16bf v; unsigned short u[16]; uint4 q[2]; }; };

__device__ __forceinline__ unsigned short f32_to_bf16(float f) {
  union { float f; unsigned int u; } c; c.f = f;
  unsigned int u = c.u;
  unsigned int r = u + 0x7FFFu + ((u >> 16) & 1u);   // RNE
  return (unsigned short)(r >> 16);
}

// per-lane 16B global -> LDS copy
__device__ __forceinline__ void copy16_g2l(const uint4* __restrict__ g, uint4* l) {
#if GEMM_ASYNC
  unsigned lo = (unsigned)(size_t)l;      // LDS byte offset = addr[31:0] (ISA 10.2)
  asm volatile("global_load_async_to_lds_b128 %0, %1, off"
               :: "v"(lo), "v"(g) : "memory");
#else
  *l = *g;
#endif
}

__device__ __forceinline__ void wait_async_copies() {
#if GEMM_ASYNC
  asm volatile("s_wait_asynccnt 0x0" ::: "memory");
#endif
}

// ---------------- conversion kernels ----------------
__global__ __launch_bounds__(256) void cvt_bf16_kernel(const float* __restrict__ s,
                                                       unsigned short* __restrict__ d,
                                                       size_t n) {
  size_t i = (size_t)blockIdx.x * 256 + threadIdx.x;
  if (i < n) d[i] = f32_to_bf16(s[i]);
}

// Transpose-convert: src f32 [K][N] -> dst bf16 [N][K].  K,N multiples of 32.
__global__ void cvtT_kernel(const float* __restrict__ src,
                            unsigned short* __restrict__ dst, int K, int N) {
  __shared__ float tile[32][33];
  int k0 = blockIdx.x * 32, n0 = blockIdx.y * 32;
  int tx = threadIdx.x, ty = threadIdx.y;      // block (32, 8)
#pragma unroll
  for (int i = 0; i < 4; ++i) {
    int k = ty + i * 8;
    tile[k][tx] = src[(size_t)(k0 + k) * N + n0 + tx];
  }
  __syncthreads();
#pragma unroll
  for (int i = 0; i < 4; ++i) {
    int n = ty + i * 8;
    dst[(size_t)(n0 + n) * K + k0 + tx] = f32_to_bf16(tile[tx][n]);
  }
}

// Pack Wq/Wk/Wv [L,H,E,DH] -> bf16 TRANSPOSED [L, 3E, E]: row col=(sel,h,d),
// inner dim e.  Writes coalesced over e.
__global__ __launch_bounds__(256) void pack_qkvT_kernel(const float* __restrict__ Wq,
                                                        const float* __restrict__ Wk,
                                                        const float* __restrict__ Wv,
                                                        unsigned short* __restrict__ dst) {
  size_t n = (size_t)LL * E3 * EE;
  size_t i = (size_t)blockIdx.x * 256 + threadIdx.x;
  if (i >= n) return;
  int e = (int)(i % EE);
  size_t rem = i / EE;
  int col = (int)(rem % E3);
  int l = (int)(rem / E3);
  int sel = col >> 10;          // 0=q 1=k 2=v
  int hd  = col & (EE - 1);
  int h   = hd >> 6;
  int d   = hd & (DHH - 1);
  const float* W = (sel == 0) ? Wq : ((sel == 1) ? Wk : Wv);
  dst[i] = f32_to_bf16(W[(((size_t)l * HH + h) * EE + e) * DHH + d]);
}

__global__ __launch_bounds__(256) void embed_kernel(const int* __restrict__ idx,
                                                    const float* __restrict__ tok,
                                                    const float* __restrict__ pos,
                                                    float* __restrict__ xf,
                                                    unsigned short* __restrict__ xb) {
  int row = blockIdx.x;                 // 0..BT-1
  int t   = row & (TT - 1);
  int id  = idx[row];
  const float* tr = tok + (size_t)id * EE;
  const float* pr = pos + (size_t)t * EE;
  float* xo = xf + (size_t)row * EE;
  unsigned short* bo = xb + (size_t)row * EE;
#pragma unroll
  for (int i = 0; i < 4; ++i) {
    int c = threadIdx.x + (i << 8);
    float v = tr[c] + pr[c];
    xo[c] = v;
    bo[c] = f32_to_bf16(v);
  }
}

// LayerNorm( y + res ) * s + b  -> fp32 + bf16 mirrors.  res may be null.
__global__ __launch_bounds__(256) void add_ln_kernel(const float* __restrict__ y,
                                                     const float* __restrict__ res,
                                                     const float* __restrict__ sc,
                                                     const float* __restrict__ bi,
                                                     float* __restrict__ outf,
                                                     unsigned short* __restrict__ outb) {
  __shared__ float red[256];
  int row = blockIdx.x, tid = threadIdx.x;
  const float* yr = y + (size_t)row * EE;
  const float* rr = res ? res + (size_t)row * EE : nullptr;
  float v[4];
  float s = 0.f;
#pragma unroll
  for (int i = 0; i < 4; ++i) {
    int c = tid + (i << 8);
    float t = yr[c] + (rr ? rr[c] : 0.f);
    v[i] = t; s += t;
  }
  red[tid] = s; __syncthreads();
  for (int st = 128; st > 0; st >>= 1) { if (tid < st) red[tid] += red[tid + st]; __syncthreads(); }
  float mean = red[0] * (1.f / EE);
  __syncthreads();
  float vs = 0.f;
#pragma unroll
  for (int i = 0; i < 4; ++i) { float d = v[i] - mean; vs += d * d; }
  red[tid] = vs; __syncthreads();
  for (int st = 128; st > 0; st >>= 1) { if (tid < st) red[tid] += red[tid + st]; __syncthreads(); }
  float rstd = rsqrtf(red[0] * (1.f / EE) + 1e-5f);
  float* of = outf + (size_t)row * EE;
  unsigned short* ob = outb + (size_t)row * EE;
#pragma unroll
  for (int i = 0; i < 4; ++i) {
    int c = tid + (i << 8);
    float o = (v[i] - mean) * rstd * sc[c] + bi[c];
    of[c] = o;
    ob[c] = f32_to_bf16(o);
  }
}

// ---------------- WMMA bf16 GEMM: C[M,N] = act(A[M,K] @ Bt^T + bias) -------
// Bt is PRE-TRANSPOSED bf16 [N][K].  Block tile 128x128, K-step 32, 8 waves
// (2M x 4N), wave tile 64x32 (4x2 frags).  Double-buffered LDS; tile k+1 is
// DMA'd (global_load_async_to_lds_b128, ASYNCcnt) while tile k computes.
__global__ __launch_bounds__(256) void gemm_bf16_kernel(const unsigned short* __restrict__ A,
                                                        const unsigned short* __restrict__ Bt,
                                                        float* __restrict__ C,
                                                        const float* __restrict__ bias,
                                                        int M, int N, int K, int relu) {
  __shared__ __align__(16) unsigned short As[2][128 * 32];    // [m][k]
  __shared__ __align__(16) unsigned short Bst[2][128 * 32];   // [n][k]
  const int tid  = threadIdx.x;
  const int m0   = blockIdx.y * 128;
  const int n0   = blockIdx.x * 128;
  const int wid  = tid >> 5, lane = tid & 31;
  const int hf   = lane >> 4, r16 = lane & 15;
  const int wm   = (wid & 1) * 64, wn = (wid >> 1) * 32;

  v8f acc[4][2];
#pragma unroll
  for (int a = 0; a < 4; ++a)
#pragma unroll
    for (int b = 0; b < 2; ++b)
#pragma unroll
      for (int i = 0; i < 8; ++i) acc[a][b][i] = 0.f;

  const uint4* Ag4 = reinterpret_cast<const uint4*>(A);
  const uint4* Bg4 = reinterpret_cast<const uint4*>(Bt);

  // per-thread staging coordinates: 512 uint4 per tile, 2 per thread
  const int r0 = tid >> 2, c4 = tid & 3;       // rows r0 and r0+64

  auto stage = [&](int k0, int buf) {
    uint4* As4  = reinterpret_cast<uint4*>(As[buf]);
    uint4* Bst4 = reinterpret_cast<uint4*>(Bst[buf]);
#pragma unroll
    for (int i = 0; i < 2; ++i) {
      int r = r0 + i * 64;
      copy16_g2l(&Ag4[(((size_t)(m0 + r) * K + k0) >> 3) + c4], &As4[r * 4 + c4]);
      copy16_g2l(&Bg4[(((size_t)(n0 + r) * K + k0) >> 3) + c4], &Bst4[r * 4 + c4]);
    }
  };

  const int nk = K >> 5;
  stage(0, 0);
  wait_async_copies();
  __syncthreads();

  for (int kt = 0; kt < nk; ++kt) {
    const int cur = kt & 1;
    const bool more = (kt + 1) < nk;
    if (more) stage((kt + 1) << 5, cur ^ 1);   // overlap with compute below

    BF16Frag a[4], b2[2];
#pragma unroll
    for (int mi = 0; mi < 4; ++mi) {
      const unsigned short* ar = &As[cur][(wm + mi * 16 + r16) * 32 + hf * 8];
      a[mi].q[0] = *reinterpret_cast<const uint4*>(ar);
      a[mi].q[1] = *reinterpret_cast<const uint4*>(ar + 16);
    }
#pragma unroll
    for (int ni = 0; ni < 2; ++ni) {
      const unsigned short* br = &Bst[cur][(wn + ni * 16 + r16) * 32 + hf * 8];
      b2[ni].q[0] = *reinterpret_cast<const uint4*>(br);
      b2[ni].q[1] = *reinterpret_cast<const uint4*>(br + 16);
    }
#pragma unroll
    for (int mi = 0; mi < 4; ++mi)
#pragma unroll
      for (int ni = 0; ni < 2; ++ni)
        acc[mi][ni] = __builtin_amdgcn_wmma_f32_16x16x32_bf16(
            false, a[mi].v, false, b2[ni].v, (short)0, acc[mi][ni], false, false);

    if (more) wait_async_copies();
    __syncthreads();
  }

#pragma unroll
  for (int mi = 0; mi < 4; ++mi)
#pragma unroll
    for (int ni = 0; ni < 2; ++ni)
#pragma unroll
      for (int i = 0; i < 8; ++i) {
        int row = m0 + wm + mi * 16 + hf * 8 + i;
        int col = n0 + wn + ni * 16 + r16;
        float v = acc[mi][ni][i];
        if (bias) v += bias[col];
        if (relu) v = fmaxf(v, 0.f);
        C[(size_t)row * N + col] = v;
      }
}

// ---------------- flash attention (causal, softmax online) ----------------
// qkv: bf16 [B, T, 3E] (q | k | v, each h*DH+d).  Out: fp32 [B, T, E].
// Block = 4 waves, each wave owns 16 query rows; key blocks of 32 in LDS.
// K tile [key][d] (B-frag reads contiguous); V tile stored TRANSPOSED
// [d][key] so the P x V B-fragment reads are contiguous too.
__global__ __launch_bounds__(128) void attention_kernel(const unsigned short* __restrict__ qkv,
                                                        float* __restrict__ Og) {
  __shared__ __align__(16) unsigned short Kt[32 * 64];     // [key][d]
  __shared__ __align__(16) unsigned short Vtt[64 * 32];    // [d][key]
  __shared__ __align__(16) unsigned short Pl[4][16 * 32];
  const int tid = threadIdx.x;
  const int bh  = blockIdx.y;
  const int b   = bh >> 4, h = bh & 15;
  const int qwg = blockIdx.x * 64;
  const int wid = tid >> 5, lane = tid & 31;
  const int hf  = lane >> 4, r16 = lane & 15;
  const int q0  = qwg + wid * 16;
  const unsigned short* base = qkv + (size_t)b * TT * E3;
  const uint4* base4 = reinterpret_cast<const uint4*>(base);
  const uint2* base2 = reinterpret_cast<const uint2*>(base);

  // Q fragments (A-layout), 16 rows x 64 dims = 2 K-chunks of 32
  BF16Frag aq[2];
#pragma unroll
  for (int c = 0; c < 2; ++c) {
    const unsigned short* qr = base + (size_t)(q0 + r16) * E3 + h * DHH + c * 32;
    aq[c].q[0] = *reinterpret_cast<const uint4*>(qr + hf * 8);
    aq[c].q[1] = *reinterpret_cast<const uint4*>(qr + 16 + hf * 8);
  }

  v8f Ofr[4];
  float m_i[8], l_i[8];
#pragma unroll
  for (int j = 0; j < 4; ++j)
#pragma unroll
    for (int i = 0; i < 8; ++i) Ofr[j][i] = 0.f;
#pragma unroll
  for (int i = 0; i < 8; ++i) { m_i[i] = -INFINITY; l_i[i] = 0.f; }

  uint4* K4lds = reinterpret_cast<uint4*>(Kt);
  unsigned int* V32 = reinterpret_cast<unsigned int*>(Vtt);

  const int sMax = (qwg + 63) >> 5;
  for (int s = 0; s <= sMax; ++s) {
    // K tile (32x64 bf16): 256 uint4, b128 both sides
#pragma unroll
    for (int i2 = 0; i2 < 2; ++i2) {
      int idx = tid + (i2 << 7);
      int r = idx >> 3, c4 = idx & 7;
      size_t ks = (size_t)(s * 32 + r) * E3 + EE + h * DHH;
      K4lds[r * 8 + c4] = base4[(ks >> 3) + c4];
    }
    // V tile transposed: 256 (key-pair, d-quad) cells, b64 global reads
#pragma unroll
    for (int i2 = 0; i2 < 2; ++i2) {
      int idx = tid + (i2 << 7);
      int dq = idx & 15, kp = idx >> 4;          // d = dq*4, key = kp*2
      size_t vs = (size_t)(s * 32 + kp * 2) * E3 + 2 * EE + h * DHH;
      uint2 a0 = base2[(vs >> 2) + dq];
      uint2 a1 = base2[((vs + E3) >> 2) + dq];
      int d0 = dq * 4;
      V32[(d0 + 0) * 16 + kp] = (a0.x & 0xFFFFu) | (a1.x << 16);
      V32[(d0 + 1) * 16 + kp] = (a0.x >> 16) | (a1.x & 0xFFFF0000u);
      V32[(d0 + 2) * 16 + kp] = (a0.y & 0xFFFFu) | (a1.y << 16);
      V32[(d0 + 3) * 16 + kp] = (a0.y >> 16) | (a1.y & 0xFFFF0000u);
    }
    __syncthreads();

    if (s * 32 <= q0 + 15) {                   // wave-uniform
      float S[2][8];
#pragma unroll
      for (int n0i = 0; n0i < 2; ++n0i) {
        v8f sf;
#pragma unroll
        for (int i = 0; i < 8; ++i) sf[i] = 0.f;
#pragma unroll
        for (int c = 0; c < 2; ++c) {
          BF16Frag bk;                          // B = K^T : [d=32][key=16]
          const unsigned short* kr = &Kt[(n0i * 16 + r16) * 64 + c * 32 + hf * 8];
          bk.q[0] = *reinterpret_cast<const uint4*>(kr);
          bk.q[1] = *reinterpret_cast<const uint4*>(kr + 16);
          sf = __builtin_amdgcn_wmma_f32_16x16x32_bf16(
              false, aq[c].v, false, bk.v, (short)0, sf, false, false);
        }
        int kk = s * 32 + n0i * 16 + r16;
#pragma unroll
        for (int i = 0; i < 8; ++i) {
          int tr = q0 + hf * 8 + i;
          S[n0i][i] = (kk > tr) ? -INFINITY : sf[i] * 0.125f;  // DH^-0.5
        }
      }
      float alpha[8];
#pragma unroll
      for (int i = 0; i < 8; ++i) {
        float nm = fmaxf(S[0][i], S[1][i]);
        nm = fmaxf(nm, __shfl_xor(nm, 1, 32));
        nm = fmaxf(nm, __shfl_xor(nm, 2, 32));
        nm = fmaxf(nm, __shfl_xor(nm, 4, 32));
        nm = fmaxf(nm, __shfl_xor(nm, 8, 32));
        float mn = fmaxf(m_i[i], nm);
        alpha[i] = __expf(m_i[i] - mn);
        m_i[i] = mn;
      }
#pragma unroll
      for (int i = 0; i < 8; ++i) {
        float p0 = __expf(S[0][i] - m_i[i]);
        float p1 = __expf(S[1][i] - m_i[i]);
        Pl[wid][(hf * 8 + i) * 32 + r16]      = f32_to_bf16(p0);
        Pl[wid][(hf * 8 + i) * 32 + 16 + r16] = f32_to_bf16(p1);
        float ps = p0 + p1;
        ps += __shfl_xor(ps, 1, 32);
        ps += __shfl_xor(ps, 2, 32);
        ps += __shfl_xor(ps, 4, 32);
        ps += __shfl_xor(ps, 8, 32);
        l_i[i] = l_i[i] * alpha[i] + ps;
      }
#pragma unroll
      for (int j = 0; j < 4; ++j)
#pragma unroll
        for (int i = 0; i < 8; ++i) Ofr[j][i] *= alpha[i];

      BF16Frag pa;                              // P in A-layout (16x32)
      {
        const unsigned short* pr = &Pl[wid][r16 * 32 + hf * 8];
        pa.q[0] = *reinterpret_cast<const uint4*>(pr);
        pa.q[1] = *reinterpret_cast<const uint4*>(pr + 16);
      }
#pragma unroll
      for (int j = 0; j < 4; ++j) {
        BF16Frag bv;                            // B = V : [key=32][d=16]
        const unsigned short* vr = &Vtt[(j * 16 + r16) * 32 + hf * 8];
        bv.q[0] = *reinterpret_cast<const uint4*>(vr);
        bv.q[1] = *reinterpret_cast<const uint4*>(vr + 16);
        Ofr[j] = __builtin_amdgcn_wmma_f32_16x16x32_bf16(
            false, pa.v, false, bv.v, (short)0, Ofr[j], false, false);
      }
    }
    __syncthreads();
  }

#pragma unroll
  for (int j = 0; j < 4; ++j)
#pragma unroll
    for (int i = 0; i < 8; ++i) {
      int tr = q0 + hf * 8 + i;
      Og[((size_t)b * TT + tr) * EE + h * DHH + j * 16 + r16] = Ofr[j][i] / l_i[i];
    }
}

// ---------------- loss ----------------
__global__ void zero_kernel(float* p) { if (threadIdx.x == 0 && blockIdx.x == 0) *p = 0.f; }

__global__ __launch_bounds__(256) void loss_kernel(const float* __restrict__ logits,
                                                   const int* __restrict__ targets,
                                                   float* __restrict__ loss) {
  __shared__ float red[256];
  int row = blockIdx.x, tid = threadIdx.x;
  const float* lr = logits + (size_t)row * VV;
  float mx = -INFINITY;
  for (int c = tid; c < VV; c += 256) mx = fmaxf(mx, lr[c]);
  red[tid] = mx; __syncthreads();
  for (int st = 128; st > 0; st >>= 1) { if (tid < st) red[tid] = fmaxf(red[tid], red[tid + st]); __syncthreads(); }
  mx = red[0]; __syncthreads();
  float s = 0.f;
  for (int c = tid; c < VV; c += 256) s += __expf(lr[c] - mx);
  red[tid] = s; __syncthreads();
  for (int st = 128; st > 0; st >>= 1) { if (tid < st) red[tid] += red[tid + st]; __syncthreads(); }
  if (tid == 0) {
    int t = targets[row];
    float lp = lr[t] - mx - __logf(red[0]);
    atomicAdd(loss, -lp * (1.f / BT));
  }
}

// ---------------- host ----------------
static inline unsigned cdiv256(size_t n) { return (unsigned)((n + 255) / 256); }

extern "C" void kernel_launch(void* const* d_in, const int* in_sizes, int n_in,
                              void* d_out, int out_size, void* d_ws, size_t ws_size,
                              hipStream_t stream) {
  (void)in_sizes; (void)n_in; (void)out_size; (void)ws_size;
  const int*   idx     = (const int*)d_in[0];
  const int*   targets = (const int*)d_in[1];
  const float* tok_emb = (const float*)d_in[2];
  const float* pos_emb = (const float*)d_in[3];
  const float* Wq      = (const float*)d_in[4];
  const float* Wk      = (const float*)d_in[5];
  const float* Wv      = (const float*)d_in[6];
  const float* Wo      = (const float*)d_in[7];
  const float* bo      = (const float*)d_in[8];
  const float* ln1_s   = (const float*)d_in[9];
  const float* ln1_b   = (const float*)d_in[10];
  const float* W1      = (const float*)d_in[11];
  const float* b1      = (const float*)d_in[12];
  const float* W2      = (const float*)d_in[13];
  const float* b2      = (const float*)d_in[14];
  const float* ln2_s   = (const float*)d_in[15];
  const float* ln2_b   = (const float*)d_in[16];
  const float* lnf_s   = (const float*)d_in[17];
  const float* lnf_b   = (const float*)d_in[18];
  const float* Wh      = (const float*)d_in[19];
  const float* bh      = (const float*)d_in[20];

  float* logits = (float*)d_out;
  float* loss   = logits + (size_t)BT * VV;

  char* ws = (char*)d_ws;
  size_t off = 0;
  auto alloc = [&](size_t bytes) -> void* {
    void* p = ws + off;
    off += (bytes + 255) & ~(size_t)255;
    return p;
  };
  float* x_f    = (float*)alloc((size_t)BT * EE * 4);
  float* y_f    = (float*)alloc((size_t)BT * EE * 4);
  float* qkv_f  = (float*)alloc((size_t)BT * E3 * 4);
  float* ff1_f  = (float*)alloc((size_t)BT * FEE * 4);
  float* attn_f = (float*)alloc((size_t)BT * EE * 4);
  unsigned short* xb    = (unsigned short*)alloc((size_t)BT * EE * 2);
  unsigned short* qkvb  = (unsigned short*)alloc((size_t)BT * E3 * 2);
  unsigned short* attnb = (unsigned short*)alloc((size_t)BT * EE * 2);
  unsigned short* ff1b  = (unsigned short*)alloc((size_t)BT * FEE * 2);
  unsigned short* Wqkvb = (unsigned short*)alloc((size_t)LL * E3 * EE * 2);  // [3E][E]
  unsigned short* Wob   = (unsigned short*)alloc((size_t)LL * EE * EE * 2);  // [E][E]^T
  unsigned short* W1b   = (unsigned short*)alloc((size_t)LL * FEE * EE * 2); // [4E][E]
  unsigned short* W2b   = (unsigned short*)alloc((size_t)LL * EE * FEE * 2); // [E][4E]
  unsigned short* Whb   = (unsigned short*)alloc((size_t)VV * EE * 2);       // [V][E]

  // --- weight conversion to transposed bf16 [N][K] (every launch) ---
  pack_qkvT_kernel<<<cdiv256((size_t)LL * E3 * EE), 256, 0, stream>>>(Wq, Wk, Wv, Wqkvb);
  for (int l = 0; l < LL; ++l) {
    cvtT_kernel<<<dim3(EE / 32, EE / 32), dim3(32, 8), 0, stream>>>(
        Wo + (size_t)l * EE * EE, Wob + (size_t)l * EE * EE, EE, EE);
    cvtT_kernel<<<dim3(EE / 32, FEE / 32), dim3(32, 8), 0, stream>>>(
        W1 + (size_t)l * EE * FEE, W1b + (size_t)l * FEE * EE, EE, FEE);
    cvtT_kernel<<<dim3(FEE / 32, EE / 32), dim3(32, 8), 0, stream>>>(
        W2 + (size_t)l * FEE * EE, W2b + (size_t)l * EE * FEE, FEE, EE);
  }
  cvtT_kernel<<<dim3(EE / 32, VV / 32), dim3(32, 8), 0, stream>>>(Wh, Whb, EE, VV);

  // --- embedding ---
  embed_kernel<<<BT, 256, 0, stream>>>(idx, tok_emb, pos_emb, x_f, xb);

  // --- transformer layers ---
  for (int l = 0; l < LL; ++l) {
    gemm_bf16_kernel<<<dim3(E3 / 128, BT / 128), 256, 0, stream>>>(
        xb, Wqkvb + (size_t)l * E3 * EE, qkv_f, nullptr, BT, E3, EE, 0);
    cvt_bf16_kernel<<<cdiv256((size_t)BT * E3), 256, 0, stream>>>(qkv_f, qkvb, (size_t)BT * E3);

    attention_kernel<<<dim3(TT / 64, BB * HH), 128, 0, stream>>>(qkvb, attn_f);
    cvt_bf16_kernel<<<cdiv256((size_t)BT * EE), 256, 0, stream>>>(attn_f, attnb, (size_t)BT * EE);

    gemm_bf16_kernel<<<dim3(EE / 128, BT / 128), 256, 0, stream>>>(
        attnb, Wob + (size_t)l * EE * EE, y_f, bo + (size_t)l * EE, BT, EE, EE, 0);
    add_ln_kernel<<<BT, 256, 0, stream>>>(y_f, x_f, ln1_s + (size_t)l * EE,
                                          ln1_b + (size_t)l * EE, x_f, xb);

    gemm_bf16_kernel<<<dim3(FEE / 128, BT / 128), 256, 0, stream>>>(
        xb, W1b + (size_t)l * FEE * EE, ff1_f, b1 + (size_t)l * FEE, BT, FEE, EE, 1);
    cvt_bf16_kernel<<<cdiv256((size_t)BT * FEE), 256, 0, stream>>>(ff1_f, ff1b, (size_t)BT * FEE);

    gemm_bf16_kernel<<<dim3(EE / 128, BT / 128), 256, 0, stream>>>(
        ff1b, W2b + (size_t)l * EE * FEE, y_f, b2 + (size_t)l * EE, BT, EE, FEE, 0);
    add_ln_kernel<<<BT, 256, 0, stream>>>(y_f, x_f, ln2_s + (size_t)l * EE,
                                          ln2_b + (size_t)l * EE, x_f, xb);
  }

  // --- final LN + logits + loss ---
  add_ln_kernel<<<BT, 256, 0, stream>>>(x_f, nullptr, lnf_s, lnf_b, x_f, xb);
  gemm_bf16_kernel<<<dim3(VV / 128, BT / 128), 256, 0, stream>>>(
      xb, Whb, logits, bh, BT, VV, EE, 0);
  zero_kernel<<<1, 1, 0, stream>>>(loss);
  loss_kernel<<<BT, 256, 0, stream>>>(logits, targets, loss);
}
